// Wav2VecFeats_69638599737667
// MI455X (gfx1250) — compile-verified, matrix operand order checked by reading, not run on hardware
//
#include <hip/hip_runtime.h>
#include <hip/hip_bf16.h>

typedef __attribute__((ext_vector_type(16))) __bf16 v16bf;
typedef __attribute__((ext_vector_type(8)))  __bf16 v8bf;
typedef __attribute__((ext_vector_type(8)))  float  v8f;

union FragBF {
    v16bf v;
    v8bf  h[2];
};

__device__ __forceinline__ float gelu_tanh(float x) {
    // jax.nn.gelu default (approximate=True): 0.5*x*(1+tanh(sqrt(2/pi)*(x+0.044715*x^3)))
    float u = 0.7978845608028654f * (x + 0.044715f * x * x * x);
    return 0.5f * x * (1.0f + tanhf(u));
}

// ---------------------------------------------------------------------------
// Repack conv weights (C_out, C_in, k) fp32 -> W'[c_out][kk*512 + c_in] bf16
// ---------------------------------------------------------------------------
__global__ void repack_w_kernel(const float* __restrict__ w, __bf16* __restrict__ wq, int k) {
    int idx = blockIdx.x * blockDim.x + threadIdx.x;
    int total = 512 * 512 * k;
    if (idx >= total) return;
    int co  = idx / (512 * k);
    int rem = idx - co * 512 * k;
    int ci  = rem / k;
    int kk  = rem - ci * k;
    wq[(size_t)co * (512 * k) + kk * 512 + ci] = (__bf16)w[idx];
}

// ---------------------------------------------------------------------------
// conv0 (cin=1, k=10, s=5) + per-(b,c) instance norm over time + GELU.
// One block per (b,c). Two passes: stats (sum, sumsq) then recompute+store.
// Output layout: h0[b][t][c] bf16 (time-major, C contiguous).
// ---------------------------------------------------------------------------
__global__ void __launch_bounds__(256)
conv0_gn_gelu_kernel(const float* __restrict__ x, const float* __restrict__ w0,
                     const float* __restrict__ gn_scale, const float* __restrict__ gn_bias,
                     __bf16* __restrict__ h0) {
    const int T = 7999;
    int b = blockIdx.x >> 9;        // /512
    int c = blockIdx.x & 511;
    const float* xb = x + (size_t)b * 40000;

    float w[10];
#pragma unroll
    for (int j = 0; j < 10; j++) w[j] = w0[c * 10 + j];

    float s = 0.0f, s2 = 0.0f;
    for (int t = threadIdx.x; t < T; t += 256) {
        const float* xp = xb + 5 * t;
        float y = 0.0f;
#pragma unroll
        for (int j = 0; j < 10; j++) y += xp[j] * w[j];
        s += y;
        s2 += y * y;
    }

    __shared__ float rs[256];
    __shared__ float rs2[256];
    rs[threadIdx.x] = s;
    rs2[threadIdx.x] = s2;
    __syncthreads();
    for (int off = 128; off > 0; off >>= 1) {
        if (threadIdx.x < (unsigned)off) {
            rs[threadIdx.x]  += rs[threadIdx.x + off];
            rs2[threadIdx.x] += rs2[threadIdx.x + off];
        }
        __syncthreads();
    }
    __shared__ float sh_mu, sh_rstd;
    if (threadIdx.x == 0) {
        float mu  = rs[0] / (float)T;
        float var = rs2[0] / (float)T - mu * mu;
        sh_mu   = mu;
        sh_rstd = rsqrtf(var + 1e-5f);
    }
    __syncthreads();

    float mu   = sh_mu;
    float mult = sh_rstd * gn_scale[c];
    float add  = gn_bias[c];
    __bf16* h0b = h0 + (size_t)b * T * 512;

    for (int t = threadIdx.x; t < T; t += 256) {
        const float* xp = xb + 5 * t;
        float y = 0.0f;
#pragma unroll
        for (int j = 0; j < 10; j++) y += xp[j] * w[j];
        float z = (y - mu) * mult + add;
        h0b[(size_t)t * 512 + c] = (__bf16)gelu_tanh(z);
    }
}

// ---------------------------------------------------------------------------
// GEMM conv (C_in=512 -> C_out=512, stride 2) with bf16 WMMA, f32 accumulate,
// fused GELU, bf16 output. in: [b][t][512] bf16; wq: [c_out][Ktot=k*512] bf16.
//
// Register-blocked: each wave owns a 32x32 output tile (2x2 accumulators).
// Per K-step of 32: 2 A-frags + 2 B-frags (8x global_load_b128) -> 4 WMMAs,
// i.e. 2 loads per v_wmma_f32_16x16x32_bf16 with full register-level reuse.
// Block = 256 threads = 8 waves arranged 2(M) x 4(N): 64x128 block tile.
// ---------------------------------------------------------------------------
__global__ void __launch_bounds__(256)
convgemm_wmma_kernel(const __bf16* __restrict__ in, const __bf16* __restrict__ wq,
                     __bf16* __restrict__ out, int Tin, int Tout, int Ktot) {
    const int STRIDE = 2;
    int wave = threadIdx.x >> 5;
    int lane = threadIdx.x & 31;
    int half = lane >> 4;            // K sub-offset selector (0 -> +0, 1 -> +8)
    int lr   = lane & 15;            // A row / B column within a 16-wide frag

    int mw = wave & 1;               // 2 wave-rows  (M)
    int nw = wave >> 1;              // 4 wave-cols  (N)
    int m0 = blockIdx.x * 64 + mw * 32;    // c_out base of this wave's 32 rows
    int n0 = blockIdx.y * 128 + nw * 32;   // time  base of this wave's 32 cols
    int b  = blockIdx.z;

    const __bf16* inb  = in  + (size_t)b * Tin  * 512;
    __bf16*       outb = out + (size_t)b * Tout * 512;

    int row0 = m0 + lr;
    int row1 = m0 + 16 + lr;
    int col0 = n0 + lr;
    int col1 = n0 + 16 + lr;
    int t0 = col0 < Tout ? col0 : (Tout - 1);   // clamp keeps loads in-bounds
    int t1 = col1 < Tout ? col1 : (Tout - 1);

    const __bf16* arow0 = wq + (size_t)row0 * Ktot + half * 8;
    const __bf16* arow1 = wq + (size_t)row1 * Ktot + half * 8;

    v8f acc00 = {}, acc01 = {}, acc10 = {}, acc11 = {};

    for (int kb = 0; kb < Ktot; kb += 32) {
        FragBF a0, a1, b0, b1;
        // A fragments: two 16B chunks per 16-row weight frag (ISA 7.12.2 layout)
        a0.h[0] = *(const v8bf*)(arow0 + kb);
        a0.h[1] = *(const v8bf*)(arow0 + kb + 16);
        a1.h[0] = *(const v8bf*)(arow1 + kb);
        a1.h[1] = *(const v8bf*)(arow1 + kb + 16);
        // Prefetch next weight chunk (global_prefetch_b8)
        __builtin_prefetch((const void*)(arow0 + kb + 64), 0, 1);

        // B fragments: K = kk*512 + c_in -> contiguous c_in run at time 2*t+kk
        int kk = kb >> 9;                 // tap index; 32-chunk never crosses taps
        int c0 = (kb & 511) + half * 8;   // c_in base for this half-wave
        const __bf16* bp0 = inb + ((size_t)(STRIDE * t0 + kk) * 512 + c0);
        const __bf16* bp1 = inb + ((size_t)(STRIDE * t1 + kk) * 512 + c0);
        b0.h[0] = *(const v8bf*)(bp0);
        b0.h[1] = *(const v8bf*)(bp0 + 16);
        b1.h[0] = *(const v8bf*)(bp1);
        b1.h[1] = *(const v8bf*)(bp1 + 16);

        acc00 = __builtin_amdgcn_wmma_f32_16x16x32_bf16(false, a0.v, false, b0.v,
                                                        (short)0, acc00, false, false);
        acc01 = __builtin_amdgcn_wmma_f32_16x16x32_bf16(false, a0.v, false, b1.v,
                                                        (short)0, acc01, false, false);
        acc10 = __builtin_amdgcn_wmma_f32_16x16x32_bf16(false, a1.v, false, b0.v,
                                                        (short)0, acc10, false, false);
        acc11 = __builtin_amdgcn_wmma_f32_16x16x32_bf16(false, a1.v, false, b1.v,
                                                        (short)0, acc11, false, false);
    }

    // D layout: lane holds column N=lr, rows M = half*8 + v -> each tile stores
    // as one aligned 16B bf16 vector per lane, fused with GELU.
    int mlo = m0 + half * 8;
    int mhi = m0 + 16 + half * 8;
    if (col0 < Tout) {
        v8bf o0, o2;
#pragma unroll
        for (int v = 0; v < 8; v++) o0[v] = (__bf16)gelu_tanh(acc00[v]);
#pragma unroll
        for (int v = 0; v < 8; v++) o2[v] = (__bf16)gelu_tanh(acc10[v]);
        *(v8bf*)(outb + (size_t)col0 * 512 + mlo) = o0;
        *(v8bf*)(outb + (size_t)col0 * 512 + mhi) = o2;
    }
    if (col1 < Tout) {
        v8bf o1, o3;
#pragma unroll
        for (int v = 0; v < 8; v++) o1[v] = (__bf16)gelu_tanh(acc01[v]);
#pragma unroll
        for (int v = 0; v < 8; v++) o3[v] = (__bf16)gelu_tanh(acc11[v]);
        *(v8bf*)(outb + (size_t)col1 * 512 + mlo) = o1;
        *(v8bf*)(outb + (size_t)col1 * 512 + mhi) = o3;
    }
}

// ---------------------------------------------------------------------------
// Mean-pool over T=124 + projection (512 -> 256) + ReLU. One block per batch.
// ---------------------------------------------------------------------------
__global__ void __launch_bounds__(256)
pool_proj_kernel(const __bf16* __restrict__ h6, const float* __restrict__ proj_w,
                 const float* __restrict__ proj_b, float* __restrict__ featp) {
    const int T = 124;
    int b = blockIdx.x;
    __shared__ float feat[512];
    const __bf16* hb = h6 + (size_t)b * T * 512;

    for (int c = threadIdx.x; c < 512; c += 256) {
        float sm = 0.0f;
        for (int t = 0; t < T; t++) sm += (float)hb[(size_t)t * 512 + c];
        feat[c] = sm / (float)T;
    }
    __syncthreads();

    int p = threadIdx.x;   // PROJ == 256 == blockDim
    float sm = proj_b[p];
    const float* pw = proj_w + (size_t)p * 512;
    for (int c = 0; c < 512; c++) sm += feat[c] * pw[c];
    featp[b * 256 + p] = fmaxf(sm, 0.0f);
}

// ---------------------------------------------------------------------------
// Per-sample classifier: h1 = relu(featp @ w1[id] + b1[id]); out = h1 @ w2[id] + b2[id]
// One block (128 threads) per batch element.
// ---------------------------------------------------------------------------
__global__ void __launch_bounds__(128)
classifier_kernel(const float* __restrict__ featp, const int* __restrict__ word_ids,
                  const float* __restrict__ cls_w1, const float* __restrict__ cls_b1,
                  const float* __restrict__ cls_w2, const float* __restrict__ cls_b2,
                  float* __restrict__ out) {
    const int PROJ = 256, HID = 128, NCLS = 5;
    int b  = blockIdx.x;
    int id = word_ids[b];

    __shared__ float fp[PROJ];
    __shared__ float h1[HID];
    for (int p = threadIdx.x; p < PROJ; p += 128) fp[p] = featp[b * PROJ + p];
    __syncthreads();

    {
        int h = threadIdx.x;  // HID == 128 == blockDim
        const float* w1b = cls_w1 + (size_t)id * PROJ * HID;
        float sm = cls_b1[id * HID + h];
        for (int p = 0; p < PROJ; p++) sm += fp[p] * w1b[p * HID + h];
        h1[h] = fmaxf(sm, 0.0f);
    }
    __syncthreads();

    if (threadIdx.x < NCLS) {
        int o = threadIdx.x;
        const float* w2b = cls_w2 + (size_t)id * HID * NCLS;
        float sm = cls_b2[id * NCLS + o];
        for (int h = 0; h < HID; h++) sm += h1[h] * w2b[h * NCLS + o];
        out[b * NCLS + o] = sm;
    }
}

// ---------------------------------------------------------------------------
// Host launcher
// ---------------------------------------------------------------------------
extern "C" void kernel_launch(void* const* d_in, const int* in_sizes, int n_in,
                              void* d_out, int out_size, void* d_ws, size_t ws_size,
                              hipStream_t stream) {
    (void)in_sizes; (void)n_in; (void)out_size; (void)ws_size;

    const float* x        = (const float*)d_in[0];
    const int*   word_ids = (const int*)d_in[1];
    const float* gn_scale = (const float*)d_in[2];
    const float* gn_bias  = (const float*)d_in[3];
    const float* proj_w   = (const float*)d_in[4];
    const float* proj_b   = (const float*)d_in[5];
    const float* cls_w1   = (const float*)d_in[6];
    const float* cls_b1   = (const float*)d_in[7];
    const float* cls_w2   = (const float*)d_in[8];
    const float* cls_b2   = (const float*)d_in[9];
    const float* convw[7];
    for (int i = 0; i < 7; i++) convw[i] = (const float*)d_in[10 + i];

    static const int T[7] = {7999, 3999, 1999, 999, 499, 249, 124};
    static const int K[7] = {10, 3, 3, 3, 3, 2, 2};
    const int B = 8;

    // Carve workspace (bf16 activations + repacked bf16 weights + featp)
    char* p = (char*)d_ws;
    auto alloc = [&](size_t bytes) -> char* {
        char* r = p;
        p += (bytes + 255) & ~(size_t)255;
        return r;
    };
    __bf16* h[7];
    for (int i = 0; i < 7; i++) h[i] = (__bf16*)alloc((size_t)T[i] * 512 * 2 * B);
    __bf16* wq[7];
    for (int i = 1; i < 7; i++) wq[i] = (__bf16*)alloc((size_t)512 * 512 * K[i] * 2);
    float* featp = (float*)alloc((size_t)B * 256 * 4);

    // 1) Repack conv1..conv6 weights to bf16, tap-major K layout
    for (int i = 1; i < 7; i++) {
        int tot = 512 * 512 * K[i];
        repack_w_kernel<<<(tot + 255) / 256, 256, 0, stream>>>(convw[i], wq[i], K[i]);
    }

    // 2) conv0 + instance-norm + GELU (fp32 math, bf16 output)
    conv0_gn_gelu_kernel<<<B * 512, 256, 0, stream>>>(x, convw[0], gn_scale, gn_bias, h[0]);

    // 3) conv1..conv6 as bf16 WMMA GEMMs with fused GELU (64x128 block tiles)
    for (int i = 1; i < 7; i++) {
        dim3 grid(8, (T[i] + 127) / 128, B);
        convgemm_wmma_kernel<<<grid, 256, 0, stream>>>(h[i - 1], wq[i], h[i],
                                                       T[i - 1], T[i], 512 * K[i]);
    }

    // 4) mean pool + projection + ReLU
    pool_proj_kernel<<<B, 256, 0, stream>>>(h[6], proj_w, proj_b, featp);

    // 5) per-word classifier -> d_out (B x 5 fp32)
    classifier_kernel<<<B, 128, 0, stream>>>(featp, word_ids, cls_w1, cls_b1,
                                             cls_w2, cls_b2, (float*)d_out);
}
